// MiniTemporalOrbitalTransformer_90795608637612
// MI455X (gfx1250) — matrix operand order repaired and checked by reading, not compiled, not run on hardware
//
#include <hip/hip_runtime.h>
#include <hip/hip_bf16.h>

typedef __attribute__((ext_vector_type(16))) _Float16 v16h;
typedef __attribute__((ext_vector_type(8)))  _Float16 v8h;
typedef __attribute__((ext_vector_type(8)))  float    v8f;
typedef __attribute__((ext_vector_type(4)))  float    v4f;

static constexpr int kH   = 64;
static constexpr int kS   = 7;
static constexpr int kFIN = 30;
static constexpr int kNB  = 16;           // batch elements per workgroup tile
static constexpr int kT   = kNB * kS;     // 112 tokens per tile
static constexpr int kTH  = 256;          // 8 waves of 32
static constexpr int kWV  = 8;

// ---------------- LDS layout (bytes) ----------------
static constexpr int W_IH  = 0;                    // [256][64] f16  (w_ih, out-major)
static constexpr int W_HH  = W_IH  + 256*64*2;     // [256][64] f16
static constexpr int W_QKV = W_HH  + 256*64*2;     // [192][64] f16
static constexpr int W_OUT = W_QKV + 192*64*2;     // [64][64]  f16
static constexpr int W_F1  = W_OUT + 64*64*2;      // [128][64] f16
static constexpr int W_F2  = W_F1  + 128*64*2;     // [64][128] f16
static constexpr int W_E1  = W_F2  + 64*128*2;     // [32][64]  f16
static constexpr int W_E2  = W_E1  + 32*64*2;      // [64][32]  f16
static constexpr int W_M1  = W_E2  + 64*32*2;      // [32][64]  f16
static constexpr int W_M2  = W_M1  + 32*64*2;      // [64][32]  f16
static constexpr int W_INP = W_M2  + 64*32*2;      // [64][32]  f16 (w_in, K padded 30->32)
static constexpr int P_OFF = W_INP + 64*32*2;      // f32 param block

// float offsets inside the P block
static constexpr int PF_BIN=0, PF_PE=64, PF_BIHH=512, PF_BQKV=768, PF_BOUT=960,
  PF_G1=1024, PF_BE1=1088, PF_G2=1152, PF_BE2=1216, PF_EW=1280, PF_MW=1344,
  PF_KW=1408, PF_FB1=1472, PF_FB2=1600, PF_EB1=1664, PF_EB2=1696, PF_MB1=1760,
  PF_MB2=1792, PF_CW1=1856, PF_CB1=3904, PF_CW2=3936, PF_CB2=4064,
  PF_UW1=4096, PF_UB1=6144, PF_UW2=6176, PF_UB2=6208, PF_N=6240;

static constexpr int HACT_OFF = P_OFF + PF_N*4;        // [112][64] f32
static constexpr int A1_OFF   = HACT_OFF + kT*kH*4;    // [112][64] f32
static constexpr int A2_OFF   = A1_OFF   + kT*kH*4;    // [112][64] f32
static constexpr int SCR_OFF  = A2_OFF   + kT*kH*4;    // phase-overlaid scratch
static constexpr int SMEM_BYTES = SCR_OFF + kT*128*4;  // scratch max = [112][128] f32
static_assert(SMEM_BYTES <= 320*1024, "exceeds 320KB LDS per workgroup");

struct Params {
  const float *x, *w_in, *b_in, *w_ih, *w_hh, *b_ih, *b_hh, *w_qkv, *b_qkv,
              *w_out, *b_out, *g1, *be1, *g2, *be2, *ew, *mw, *kw,
              *e_w1, *e_b1, *e_w2, *e_b2, *m_w1, *m_b1, *m_w2, *m_b2,
              *f_w1, *f_b1, *f_w2, *f_b2, *c_w1, *c_b1, *c_w2, *c_b2,
              *u_w1, *u_b1, *u_w2, *u_b2;
  float* out;
  int B, ntiles;
};

__device__ __forceinline__ float sigm(float x) { return 1.f / (1.f + __expf(-x)); }

// A fragment (16x32 f16) for one K-chunk, sourced from an f32 row in LDS.
// Per ISA layout: lanes 0-15 hold rows M=0..15; lane's halves are
// K = K0+8*kh .. +7 (VGPR0..3) and K0+16+8*kh .. +7 (VGPR4..7).
__device__ __forceinline__ v16h a_frag_f32(const float* row, int k0, int kh) {
  const float* p = row + k0 + 8*kh;
  v4f r0 = *(const v4f*)(p);
  v4f r1 = *(const v4f*)(p + 4);
  v4f r2 = *(const v4f*)(p + 16);
  v4f r3 = *(const v4f*)(p + 20);
  v16h a;
  a[0]=(_Float16)r0[0];  a[1]=(_Float16)r0[1];  a[2]=(_Float16)r0[2];  a[3]=(_Float16)r0[3];
  a[4]=(_Float16)r1[0];  a[5]=(_Float16)r1[1];  a[6]=(_Float16)r1[2];  a[7]=(_Float16)r1[3];
  a[8]=(_Float16)r2[0];  a[9]=(_Float16)r2[1];  a[10]=(_Float16)r2[2]; a[11]=(_Float16)r2[3];
  a[12]=(_Float16)r3[0]; a[13]=(_Float16)r3[1]; a[14]=(_Float16)r3[2]; a[15]=(_Float16)r3[3];
  return a;
}

// B fragment (32x16 f16): weight stored out-major [N][K] (== column-major B).
// Lane n (0-15) holds col N=n, K=K0+8*kh..+7 and K0+16+8*kh..+7 (kh = lane half).
__device__ __forceinline__ v16h b_frag(const _Float16* w, int Kt, int n, int k0, int kh) {
  const _Float16* p = w + n*Kt + k0 + 8*kh;
  v8h b0 = *(const v8h*)(p);
  v8h b1 = *(const v8h*)(p + 16);
  v16h b;
  b[0]=b0[0]; b[1]=b0[1]; b[2]=b0[2]; b[3]=b0[3];
  b[4]=b0[4]; b[5]=b0[5]; b[6]=b0[6]; b[7]=b0[7];
  b[8]=b1[0]; b[9]=b1[1]; b[10]=b1[2]; b[11]=b1[3];
  b[12]=b1[4]; b[13]=b1[5]; b[14]=b1[6]; b[15]=b1[7];
  return b;
}

// Token-level GEMM: D[M=16*Mt, N=16*Nt] = A(f32 LDS) * W(f16 LDS) + bias.
// MODE: 0 = f32 store, 1 = f32 store with ReLU, 2 = f16 store.
template <int KC, int MODE>
__device__ void gemm(const float* A, int as, const _Float16* W, int Kt,
                     float* D, _Float16* D16, int ds_, const float* bias,
                     int Mt, int Nt, int wave, int lane) {
  const int ml = lane & 15, kh = lane >> 4;
  const int total = Mt * Nt;
  for (int tile = wave; tile < total; tile += kWV) {
    int mt = tile / Nt, nt = tile - mt * Nt;
    const float* arow = A + (mt*16 + ml) * as;
    int n = nt*16 + ml;
    float bv = bias[n];
    v8f c;
#pragma unroll
    for (int r = 0; r < 8; ++r) c[r] = bv;
#pragma unroll
    for (int kc = 0; kc < KC; ++kc) {
      v16h a = a_frag_f32(arow, kc*32, kh);
      v16h b = b_frag(W, Kt, n, kc*32, kh);
      c = __builtin_amdgcn_wmma_f32_16x16x32_f16(false, a, false, b, (short)0, c, false, false);
    }
    int r0 = mt*16 + kh*8;   // D: lane holds rows 8*kh+r, col n
    if (MODE == 2) {
#pragma unroll
      for (int r = 0; r < 8; ++r) D16[(r0+r)*ds_ + n] = (_Float16)c[r];
    } else if (MODE == 1) {
#pragma unroll
      for (int r = 0; r < 8; ++r) D[(r0+r)*ds_ + n] = fmaxf(c[r], 0.f);
    } else {
#pragma unroll
      for (int r = 0; r < 8; ++r) D[(r0+r)*ds_ + n] = c[r];
    }
  }
}

__global__ __launch_bounds__(kTH)
void MiniTemporalOrbitalTransformer_90795608637612_kernel(Params p) {
  extern __shared__ char smem[];
  float* P    = (float*)(smem + P_OFF);
  float* hact = (float*)(smem + HACT_OFF);
  float* a1   = (float*)(smem + A1_OFF);
  float* a2   = (float*)(smem + A2_OFF);
  char*  scr  = smem + SCR_OFF;
  const int tid = threadIdx.x;
  const int wave = tid >> 5, lane = tid & 31;

  // ---------------- one-time weight staging (f32 -> f16 LDS) ----------------
  auto cvt = [&](const float* g, int off, int n) {
    _Float16* d = (_Float16*)(smem + off);
    for (int i = tid; i < n; i += kTH) d[i] = (_Float16)g[i];
  };
  cvt(p.w_ih, W_IH, 256*64);   cvt(p.w_hh, W_HH, 256*64);
  cvt(p.w_qkv, W_QKV, 192*64); cvt(p.w_out, W_OUT, 64*64);
  cvt(p.f_w1, W_F1, 128*64);   cvt(p.f_w2, W_F2, 64*128);
  cvt(p.e_w1, W_E1, 32*64);    cvt(p.e_w2, W_E2, 64*32);
  cvt(p.m_w1, W_M1, 32*64);    cvt(p.m_w2, W_M2, 64*32);
  { _Float16* d = (_Float16*)(smem + W_INP);
    for (int i = tid; i < 64*32; i += kTH) {
      int r = i >> 5, c = i & 31;
      d[i] = (c < kFIN) ? (_Float16)p.w_in[r*kFIN + c] : (_Float16)0.f;
    } }
  auto cpf = [&](const float* g, int fo, int n) {
    for (int i = tid; i < n; i += kTH) P[fo + i] = g[i];
  };
  cpf(p.b_in, PF_BIN, 64);
  for (int i = tid; i < 256; i += kTH) P[PF_BIHH + i] = p.b_ih[i] + p.b_hh[i];
  cpf(p.b_qkv, PF_BQKV, 192); cpf(p.b_out, PF_BOUT, 64);
  cpf(p.g1, PF_G1, 64); cpf(p.be1, PF_BE1, 64);
  cpf(p.g2, PF_G2, 64); cpf(p.be2, PF_BE2, 64);
  cpf(p.ew, PF_EW, 64); cpf(p.mw, PF_MW, 64); cpf(p.kw, PF_KW, 64);
  cpf(p.f_b1, PF_FB1, 128); cpf(p.f_b2, PF_FB2, 64);
  cpf(p.e_b1, PF_EB1, 32);  cpf(p.e_b2, PF_EB2, 64);
  cpf(p.m_b1, PF_MB1, 32);  cpf(p.m_b2, PF_MB2, 64);
  cpf(p.c_w1, PF_CW1, 2048); cpf(p.c_b1, PF_CB1, 32);
  cpf(p.c_w2, PF_CW2, 128);  cpf(p.c_b2, PF_CB2, 4);
  cpf(p.u_w1, PF_UW1, 2048); cpf(p.u_b1, PF_UB1, 32);
  cpf(p.u_w2, PF_UW2, 32);   cpf(p.u_b2, PF_UB2, 1);
  // positional encoding table pe[s][d]
  for (int i = tid; i < kS*kH; i += kTH) {
    int s = i >> 6, d = i & 63, k = d >> 1;
    float div = __expf(-(float)(2*k) * (9.210340371976184f / 64.f));
    P[PF_PE + i] = (d & 1) ? __cosf((float)s * div) : __sinf((float)s * div);
  }
  __syncthreads();

  // ---------------- persistent loop over 16-batch tiles ----------------
  for (int tileIdx = blockIdx.x; tileIdx < p.ntiles; tileIdx += gridDim.x) {
    const int b0 = tileIdx * kNB;
    // prefetch next tile's input (global_prefetch_b8)
    if (tileIdx + (int)gridDim.x < p.ntiles)
      __builtin_prefetch(p.x + (size_t)(tileIdx + gridDim.x)*kNB*kS*kFIN + tid*13, 0, 3);

    // ---- stage x -> XIN [112][32] f32 via GLOBAL_LOAD_ASYNC_TO_LDS ----
    // Tile's x is fully contiguous in global memory: [b0*7*30, +3360).
    // Each of 224 threads owns a contiguous 15-element run of one row-half;
    // async ops write LDS directly (tracked by ASYNCcnt).
    float* xin = (float*)scr;
    if (tid < 2*kT) {
      int t = tid >> 1, half = tid & 1;                     // token row, half-row
      unsigned go = (unsigned)((b0*kS*kFIN + t*kFIN + half*15) * 4);
      unsigned lo = (unsigned)(SCR_OFF + (t*32 + half*15) * 4);
#pragma unroll
      for (int j = 0; j < 15; ++j) {
        asm volatile("global_load_async_to_lds_b32 %0, %1, %2 offset:0"
                     :: "v"(lo + 4u*j), "v"(go + 4u*j), "s"(p.x)
                     : "memory");
      }
    }
    if (tid < kT) { xin[tid*32 + 30] = 0.f; xin[tid*32 + 31] = 0.f; }  // K pad
    asm volatile("s_wait_asynccnt 0x0" ::: "memory");
    __syncthreads();

    // ---- input projection: hact = xin @ w_in^T + b_in ----
    gemm<1,0>(xin, 32, (const _Float16*)(smem + W_INP), 32,
              hact, nullptr, 64, P + PF_BIN, 7, 4, wave, lane);
    __syncthreads();
    for (int i = tid; i < kT*kH; i += kTH) { int t = i >> 6; hact[i] += P[PF_PE + (t%7)*64 + (i&63)]; }
    __syncthreads();

    // ---- LSTM over S=7 steps: gates = h_s @ w_ih^T + hstate @ w_hh^T + (b_ih+b_hh) ----
    float* G  = (float*)scr;            // [16][256]
    float* HS = (float*)(scr + 16384);  // [16][64]
    float* CS = (float*)(scr + 20480);  // [16][64]
    for (int i = tid; i < kNB*kH; i += kTH) { HS[i] = 0.f; CS[i] = 0.f; }
    __syncthreads();
    for (int s = 0; s < kS; ++s) {
      { const int ml = lane & 15, kh = lane >> 4;
        for (int nt = wave; nt < 16; nt += kWV) {
          int n = nt*16 + ml;
          float bv = P[PF_BIHH + n];
          v8f c;
#pragma unroll
          for (int r = 0; r < 8; ++r) c[r] = bv;
          const float* ar1 = hact + (ml*7 + s)*64;   // token row of local batch ml
          const float* ar2 = HS + ml*64;
#pragma unroll
          for (int kc = 0; kc < 2; ++kc) {
            v16h a = a_frag_f32(ar1, kc*32, kh);
            v16h b = b_frag((const _Float16*)(smem + W_IH), 64, n, kc*32, kh);
            c = __builtin_amdgcn_wmma_f32_16x16x32_f16(false, a, false, b, (short)0, c, false, false);
          }
#pragma unroll
          for (int kc = 0; kc < 2; ++kc) {
            v16h a = a_frag_f32(ar2, kc*32, kh);
            v16h b = b_frag((const _Float16*)(smem + W_HH), 64, n, kc*32, kh);
            c = __builtin_amdgcn_wmma_f32_16x16x32_f16(false, a, false, b, (short)0, c, false, false);
          }
          int r0 = kh*8;
#pragma unroll
          for (int r = 0; r < 8; ++r) G[(r0+r)*256 + n] = c[r];
        } }
      __syncthreads();
      for (int i = tid; i < kNB*kH; i += kTH) {
        int b = i >> 6, d = i & 63;
        float gi = G[b*256 + d], gf = G[b*256 + 64 + d];
        float gg = G[b*256 + 128 + d], go = G[b*256 + 192 + d];
        float cn = sigm(gf)*CS[i] + sigm(gi)*tanhf(gg);
        float hn = sigm(go)*tanhf(cn);
        CS[i] = cn; HS[i] = hn;
        a1[(b*7 + s)*64 + d] = hn;
      }
      __syncthreads();
    }

    // ---- h1 = LN(lstm_out + h) -> a1 ----
    for (int t = tid; t < kT; t += kTH) {
      float* xr = a1 + t*64; const float* hr = hact + t*64;
      float mu = 0.f;
      for (int d = 0; d < 64; ++d) mu += xr[d] + hr[d];
      mu *= (1.f/64.f);
      float var = 0.f;
      for (int d = 0; d < 64; ++d) { float v = xr[d] + hr[d] - mu; var += v*v; }
      float inv = rsqrtf(var*(1.f/64.f) + 1e-5f);
      for (int d = 0; d < 64; ++d)
        xr[d] = (xr[d] + hr[d] - mu)*inv*P[PF_G1 + d] + P[PF_BE1 + d];
    }
    __syncthreads();

    // ---- QKV = h1 @ w_qkv^T + b_qkv  (f16 in scratch) ----
    _Float16* qkv = (_Float16*)scr;    // [112][192]
    gemm<2,2>(a1, 64, (const _Float16*)(smem + W_QKV), 64,
              nullptr, qkv, 192, P + PF_BQKV, 7, 12, wave, lane);
    __syncthreads();

    // ---- attention (S=7, dh=16) -> a2 ----
    for (int task = tid; task < kNB*4*7; task += kTH) {
      int b = task / 28, rem = task % 28, hd = rem / 7, q = rem % 7;
      const _Float16* qp = qkv + (b*7 + q)*192 + hd*16;
      float sc[7], mx = -1e30f;
      for (int k = 0; k < 7; ++k) {
        const _Float16* kp = qkv + (b*7 + k)*192 + 64 + hd*16;
        float sacc = 0.f;
        for (int d = 0; d < 16; ++d) sacc += (float)qp[d] * (float)kp[d];
        sacc *= 0.25f;                       // 1/sqrt(16)
        sc[k] = sacc; mx = fmaxf(mx, sacc);
      }
      float ssum = 0.f;
      for (int k = 0; k < 7; ++k) { sc[k] = __expf(sc[k] - mx); ssum += sc[k]; }
      float inv = 1.f / ssum;
      float o[16];
      for (int d = 0; d < 16; ++d) o[d] = 0.f;
      for (int k = 0; k < 7; ++k) {
        float w = sc[k] * inv;
        const _Float16* vp = qkv + (b*7 + k)*192 + 128 + hd*16;
        for (int d = 0; d < 16; ++d) o[d] += w * (float)vp[d];
      }
      float* dst = a2 + (b*7 + q)*64 + hd*16;
      for (int d = 0; d < 16; ++d) dst[d] = o[d];
    }
    __syncthreads();

    // ---- out projection: hact = ao @ w_out^T + b_out ----
    gemm<2,0>(a2, 64, (const _Float16*)(smem + W_OUT), 64,
              hact, nullptr, 64, P + PF_BOUT, 7, 4, wave, lane);
    __syncthreads();

    // ---- h2 = LN(ao_proj + h1) -> a1 ----
    for (int t = tid; t < kT; t += kTH) {
      float* xr = a1 + t*64; const float* hr = hact + t*64;
      float mu = 0.f;
      for (int d = 0; d < 64; ++d) mu += xr[d] + hr[d];
      mu *= (1.f/64.f);
      float var = 0.f;
      for (int d = 0; d < 64; ++d) { float v = xr[d] + hr[d] - mu; var += v*v; }
      float inv = rsqrtf(var*(1.f/64.f) + 1e-5f);
      for (int d = 0; d < 64; ++d)
        xr[d] = (xr[d] + hr[d] - mu)*inv*P[PF_G2 + d] + P[PF_BE2 + d];
    }
    __syncthreads();

    // ---- FFN: mid = relu(h2 @ f_w1^T + f_b1); a2 = mid @ f_w2^T + f_b2 ----
    float* mid = (float*)scr;   // [112][128]
    gemm<2,1>(a1, 64, (const _Float16*)(smem + W_F1), 64,
              mid, nullptr, 128, P + PF_FB1, 7, 8, wave, lane);
    __syncthreads();
    gemm<4,0>(mid, 128, (const _Float16*)(smem + W_F2), 128,
              a2, nullptr, 64, P + PF_FB2, 7, 4, wave, lane);
    __syncthreads();
    for (int i = tid; i < kT*kH; i += kTH) a2[i] += a1[i];   // feat = h2 + ffn
    __syncthreads();

    // ---- physics constraints: ec -> a1, mc -> hact ----
    float* pm = (float*)scr;    // [112][32]
    gemm<2,1>(a2, 64, (const _Float16*)(smem + W_E1), 64,
              pm, nullptr, 32, P + PF_EB1, 7, 2, wave, lane);
    __syncthreads();
    gemm<1,0>(pm, 32, (const _Float16*)(smem + W_E2), 32,
              a1, nullptr, 64, P + PF_EB2, 7, 4, wave, lane);
    __syncthreads();
    gemm<2,1>(a2, 64, (const _Float16*)(smem + W_M1), 64,
              pm, nullptr, 32, P + PF_MB1, 7, 2, wave, lane);
    __syncthreads();
    gemm<1,0>(pm, 32, (const _Float16*)(smem + W_M2), 32,
              hact, nullptr, 64, P + PF_MB2, 7, 4, wave, lane);
    __syncthreads();
    for (int i = tid; i < kT*kH; i += kTH) {
      int d = i & 63;
      float f = a2[i];
      a2[i] = f + a1[i]*P[PF_EW + d] + hact[i]*P[PF_MW + d] + tanhf(f)*P[PF_KW + d];
    }
    __syncthreads();

    // ---- mean-pool over seq + heads (VALU; tiny) ----
    float* pool = (float*)scr;            // [16][64]
    float* cm   = (float*)(scr + 4096);   // [16][32]
    float* um   = (float*)(scr + 6144);   // [16][32]
    for (int i = tid; i < kNB*kH; i += kTH) {
      int b = i >> 6, d = i & 63;
      float sacc = 0.f;
      for (int s2 = 0; s2 < 7; ++s2) sacc += a2[(b*7 + s2)*64 + d];
      pool[i] = sacc * (1.f/7.f);
    }
    __syncthreads();
    for (int i = tid; i < 1024; i += kTH) {
      int which = i >> 9, r = i & 511, b = r >> 5, j = r & 31;
      const float* w = P + (which ? PF_UW1 : PF_CW1) + j*64;
      float sacc = P[(which ? PF_UB1 : PF_CB1) + j];
      for (int d = 0; d < 64; ++d) sacc += pool[b*64 + d] * w[d];
      (which ? um : cm)[b*32 + j] = fmaxf(sacc, 0.f);
    }
    __syncthreads();
    if (tid < 64) {
      int b = tid >> 2, o = tid & 3;
      float sacc = P[PF_CB2 + o];
      for (int j = 0; j < 32; ++j) sacc += cm[b*32 + j] * P[PF_CW2 + o*32 + j];
      p.out[(size_t)(b0 + b)*4 + o] = sacc;                     // risk
    } else if (tid < 80) {
      int b = tid - 64;
      float sacc = P[PF_UB2];
      for (int j = 0; j < 32; ++j) sacc += um[b*32 + j] * P[PF_UW2 + j];
      p.out[(size_t)p.B*4 + (b0 + b)] = sigm(sacc);             // unc
    }
    __syncthreads();  // protect scratch before next tile
  }
}

extern "C" void kernel_launch(void* const* d_in, const int* in_sizes, int n_in,
                              void* d_out, int out_size, void* d_ws, size_t ws_size,
                              hipStream_t stream) {
  (void)n_in; (void)out_size; (void)d_ws; (void)ws_size;
  Params p;
  p.x     = (const float*)d_in[0];
  p.w_in  = (const float*)d_in[1];  p.b_in  = (const float*)d_in[2];
  p.w_ih  = (const float*)d_in[3];  p.w_hh  = (const float*)d_in[4];
  p.b_ih  = (const float*)d_in[5];  p.b_hh  = (const float*)d_in[6];
  p.w_qkv = (const float*)d_in[7];  p.b_qkv = (const float*)d_in[8];
  p.w_out = (const float*)d_in[9];  p.b_out = (const float*)d_in[10];
  p.g1    = (const float*)d_in[11]; p.be1   = (const float*)d_in[12];
  p.g2    = (const float*)d_in[13]; p.be2   = (const float*)d_in[14];
  p.ew    = (const float*)d_in[15]; p.mw    = (const float*)d_in[16];
  p.kw    = (const float*)d_in[17];
  p.e_w1  = (const float*)d_in[18]; p.e_b1  = (const float*)d_in[19];
  p.e_w2  = (const float*)d_in[20]; p.e_b2  = (const float*)d_in[21];
  p.m_w1  = (const float*)d_in[22]; p.m_b1  = (const float*)d_in[23];
  p.m_w2  = (const float*)d_in[24]; p.m_b2  = (const float*)d_in[25];
  p.f_w1  = (const float*)d_in[26]; p.f_b1  = (const float*)d_in[27];
  p.f_w2  = (const float*)d_in[28]; p.f_b2  = (const float*)d_in[29];
  p.c_w1  = (const float*)d_in[30]; p.c_b1  = (const float*)d_in[31];
  p.c_w2  = (const float*)d_in[32]; p.c_b2  = (const float*)d_in[33];
  p.u_w1  = (const float*)d_in[34]; p.u_b1  = (const float*)d_in[35];
  p.u_w2  = (const float*)d_in[36]; p.u_b2  = (const float*)d_in[37];
  p.out   = (float*)d_out;
  p.B      = in_sizes[0] / (kS * kFIN);
  p.ntiles = p.B / kNB;

  (void)hipFuncSetAttribute(
      reinterpret_cast<const void*>(&MiniTemporalOrbitalTransformer_90795608637612_kernel),
      hipFuncAttributeMaxDynamicSharedMemorySize, SMEM_BYTES);

  int grid = p.ntiles < 1024 ? p.ntiles : 1024;
  if (grid < 1) grid = 1;
  MiniTemporalOrbitalTransformer_90795608637612_kernel
      <<<dim3(grid), dim3(kTH), SMEM_BYTES, stream>>>(p);
}
